// GCN_drug_25254407700903
// MI455X (gfx1250) — matrix operand order repaired and checked by reading, not compiled, not run on hardware
//
#include <hip/hip_runtime.h>
#include <hip/hip_bf16.h>
#include <math.h>

typedef __attribute__((ext_vector_type(16))) __bf16 v16bf;
typedef __attribute__((ext_vector_type(8)))  __bf16 v8bf;
typedef __attribute__((ext_vector_type(8)))  float  v8f;

static constexpr int ND  = 8000;
static constexpr int NC  = 16000;
static constexpr int FD  = 128;
static constexpr int NH  = 4;      // heads (drug GAT)
static constexpr int OCH = 128;
static constexpr int ED  = 160000;
static constexpr int EC  = 320000;

static inline unsigned cdiv(long a, long b) { return (unsigned)((a + b - 1) / b); }

// ---------------------------------------------------------------------------
// WMMA GEMM, register-blocked: C[M,N] (f32) = A[M,K] bf16 * BT[N,K]^T bf16 (+bias)
// Each wave computes a 64x64 output macro-tile = 4x4 WMMA 16x16 tiles.
// Block = 128 threads = 4 waves stacked along M (block covers 256xM, 64xN).
// A frag reused across 4 N-tiles, B frag across 4 M-tiles -> 8 B of L2 reads
// per output element (vs 32 B unblocked); big GEMM becomes store-BW bound.
// Requires M%64==0 handled by guard, N%64==0, K%32==0 (true for all shapes here).
// ---------------------------------------------------------------------------
__global__ __launch_bounds__(128, 1)
void wmma_gemm_nt_kernel(const __bf16* __restrict__ A, const __bf16* __restrict__ BT,
                         const float* __restrict__ bias, float* __restrict__ C,
                         int M, int N, int K) {
  const int wave = threadIdx.x >> 5;
  const int lane = threadIdx.x & 31;
  const int n0 = blockIdx.x * 64;                // 4 n-tiles
  const int m0 = (blockIdx.y * 4 + wave) * 64;   // 4 m-tiles per wave
  if (m0 >= M) return;
  const int r = lane & 15, half = lane >> 4;

  // A fragment: lane<16 holds row, K = k0+{0..15}; lane>=16: K = k0+{16..31}
  const __bf16* ap = A  + (long)(m0 + r) * K + half * 16;
  // B fragment: lane<16 holds col, K = k0+{0..15}; lane>=16: K = k0+{16..31}
  const __bf16* bp = BT + (long)(n0 + r) * K + half * 16;
  const long tstride = 16L * K;                  // stride between 16-row tiles

  v8f acc[4][4] = {};
  for (int k0 = 0; k0 < K; k0 += 32) {
    v16bf af[4], bfr[4];
#pragma unroll
    for (int i = 0; i < 4; ++i) {
      v8bf lo = *(const v8bf*)(ap + i * tstride + k0);
      v8bf hi = *(const v8bf*)(ap + i * tstride + k0 + 8);
      af[i] = __builtin_shufflevector(lo, hi, 0,1,2,3,4,5,6,7,8,9,10,11,12,13,14,15);
    }
#pragma unroll
    for (int j = 0; j < 4; ++j)
      bfr[j] = *(const v16bf*)(bp + j * tstride + k0);
#pragma unroll
    for (int i = 0; i < 4; ++i)
#pragma unroll
      for (int j = 0; j < 4; ++j)
        acc[i][j] = __builtin_amdgcn_wmma_f32_16x16x32_bf16(false, af[i], false, bfr[j],
                                                            (short)0, acc[i][j], false, false);
  }
#pragma unroll
  for (int j = 0; j < 4; ++j) {
    const int col = n0 + j * 16 + r;
    const float bv = bias ? bias[col] : 0.0f;
#pragma unroll
    for (int i = 0; i < 4; ++i) {
      const int row0 = m0 + i * 16 + half * 8;   // VGPR v -> row row0+v
#pragma unroll
      for (int v = 0; v < 8; ++v)
        C[(long)(row0 + v) * N + col] = acc[i][j][v] + bv;
    }
  }
}

// ---------------------------------------------------------------------------
// Small helper kernels
// ---------------------------------------------------------------------------
__global__ void fill_kernel(float* __restrict__ p, float v, long n) {
  long i = (long)blockIdx.x * blockDim.x + threadIdx.x;
  if (i < n) p[i] = v;
}
__global__ void cvt_bf16_kernel(const float* __restrict__ in, __bf16* __restrict__ out, long n) {
  long i = (long)blockIdx.x * blockDim.x + threadIdx.x;
  if (i < n) out[i] = (__bf16)in[i];
}
// in is [K,N] row-major f32 -> out [N,K] bf16
__global__ void transpose_bf16_kernel(const float* __restrict__ in, __bf16* __restrict__ out,
                                      int K, int N) {
  long i = (long)blockIdx.x * blockDim.x + threadIdx.x;
  if (i < (long)K * N) {
    int k = (int)(i / N), n = (int)(i % N);
    out[(long)n * K + k] = (__bf16)in[i];
  }
}
__global__ void deg_edges_kernel(const int* __restrict__ dst, const float* __restrict__ ew,
                                 float* __restrict__ deg, int E) {
  int e = blockIdx.x * blockDim.x + threadIdx.x;
  if (e < E) atomicAdd(&deg[dst[e]], ew[e]);
}
__global__ void dinv_kernel(const float* __restrict__ deg, float* __restrict__ dinv, int n) {
  int i = blockIdx.x * blockDim.x + threadIdx.x;
  if (i < n) { float d = deg[i]; dinv[i] = d > 0.0f ? rsqrtf(d) : 0.0f; }
}
__global__ void edge_norm_kernel(const int* __restrict__ src, const int* __restrict__ dst,
                                 const float* __restrict__ ew, const float* __restrict__ dinv,
                                 float* __restrict__ norm, int E) {
  int e = blockIdx.x * blockDim.x + threadIdx.x;
  if (e < E) norm[e] = dinv[src[e]] * ew[e] * dinv[dst[e]];
}
// out[i,f] = dinv[i]^2 * h[i,f] + bias[f]   (self loop, weight 1, + GCN bias)
__global__ void gcn_self_bias_kernel(const float* __restrict__ h, const float* __restrict__ dinv,
                                     const float* __restrict__ bias, float* __restrict__ out,
                                     long nF) {
  long t = (long)blockIdx.x * blockDim.x + threadIdx.x;
  if (t < nF) {
    int i = (int)(t >> 7), f = (int)(t & 127);
    float di = dinv[i];
    out[t] = di * di * h[t] + bias[f];
  }
}
__global__ void gcn_scatter_kernel(const int* __restrict__ src, const int* __restrict__ dst,
                                   const float* __restrict__ norm, const float* __restrict__ h,
                                   float* __restrict__ out, long EF) {
  long t = (long)blockIdx.x * blockDim.x + threadIdx.x;
  if (t < EF) {
    int e = (int)(t >> 7), f = (int)(t & 127);
    atomicAdd(&out[(long)dst[e] * FD + f], norm[e] * h[(long)src[e] * FD + f]);
  }
}
// Per-(node,head) dots with att_src / att_dst, one wave each
__global__ void att_sums_kernel(const float* __restrict__ h, const float* __restrict__ att_s,
                                const float* __restrict__ att_d, float* __restrict__ asum,
                                float* __restrict__ dsum, int n, int H) {
  int wid  = blockIdx.x * (blockDim.x >> 5) + (threadIdx.x >> 5);
  int lane = threadIdx.x & 31;
  if (wid >= n * H) return;
  int node = wid / H, hd = wid % H;
  const float* hp = h + (long)node * H * FD + (long)hd * FD;
  const float* sp = att_s + hd * FD;
  const float* dp = att_d + hd * FD;
  float sa = 0.f, sd = 0.f;
  for (int f = lane; f < FD; f += 32) { float v = hp[f]; sa += v * sp[f]; sd += v * dp[f]; }
  for (int m = 16; m; m >>= 1) { sa += __shfl_xor(sa, m, 32); sd += __shfl_xor(sd, m, 32); }
  if (lane == 0) { asum[wid] = sa; dsum[wid] = sd; }
}
__global__ void mean_reduce_kernel(const float* __restrict__ x, float* __restrict__ out,
                                   int n, float inv_n) {
  int i = blockIdx.x * blockDim.x + threadIdx.x;
  if (i < n) atomicAdd(out, x[i] * inv_n);
}
// we_dot[hd] = sum_f We[hd*F+f]*ae[hd*F+f]
__global__ void wedot_kernel(const float* __restrict__ We, const float* __restrict__ ae,
                             float* __restrict__ out, int H) {
  int hd = threadIdx.x;
  if (hd < H) {
    float s = 0.f;
    for (int f = 0; f < FD; ++f) s += We[hd * FD + f] * ae[hd * FD + f];
    out[hd] = s;
  }
}
__device__ inline void atomicMaxF(float* addr, float val) {
  if (val >= 0.0f) atomicMax((int*)addr, __float_as_int(val));
  else             atomicMin((unsigned int*)addr, __float_as_uint(val));
}
__global__ void alpha_kernel(const int* __restrict__ src, const int* __restrict__ dst,
                             const float* __restrict__ asum, const float* __restrict__ dsum,
                             const float* __restrict__ ea, const float* __restrict__ mean_ea,
                             const float* __restrict__ wedot, float* __restrict__ alpha,
                             float* __restrict__ mbuf, int E, int n, int H) {
  long idx = (long)blockIdx.x * blockDim.x + threadIdx.x;
  if (idx >= (long)(E + n) * H) return;
  int e = (int)(idx / H), hd = (int)(idx % H);
  int s, d;
  if (e < E) { s = src[e]; d = dst[e]; } else { s = d = e - E; }
  float a = asum[s * H + hd] + dsum[d * H + hd];
  if (ea) {
    float v = (e < E) ? ea[e] : mean_ea[0];
    a += v * wedot[hd];
  }
  a = a > 0.f ? a : 0.2f * a;          // leaky_relu 0.2
  alpha[idx] = a;
  atomicMaxF(&mbuf[d * H + hd], a);
}
__global__ void exp_kernel(const int* __restrict__ dst, const float* __restrict__ mbuf,
                           float* __restrict__ alpha, float* __restrict__ ssum,
                           int E, int n, int H) {
  long idx = (long)blockIdx.x * blockDim.x + threadIdx.x;
  if (idx >= (long)(E + n) * H) return;
  int e = (int)(idx / H), hd = (int)(idx % H);
  int d = (e < E) ? dst[e] : e - E;
  float p = __expf(alpha[idx] - mbuf[d * H + hd]);
  alpha[idx] = p;
  atomicAdd(&ssum[d * H + hd], p);
}
__global__ void init_rows_bias_kernel(float* __restrict__ out, const float* __restrict__ bias,
                                      long nF) {
  long t = (long)blockIdx.x * blockDim.x + threadIdx.x;
  if (t < nF) out[t] = bias[t & 127];
}
__global__ void gat_scatter_kernel(const int* __restrict__ src, const int* __restrict__ dst,
                                   const float* __restrict__ p, const float* __restrict__ ssum,
                                   const float* __restrict__ h, float* __restrict__ out,
                                   int E, int n, int H) {
  long t = (long)blockIdx.x * blockDim.x + threadIdx.x;
  if (t >= (long)(E + n) * FD) return;
  int e = (int)(t >> 7), f = (int)(t & 127);
  int s, d;
  if (e < E) { s = src[e]; d = dst[e]; } else { s = d = e - E; }
  float acc = 0.f;
  for (int hd = 0; hd < H; ++hd)
    acc += (p[(long)e * H + hd] / ssum[d * H + hd]) * h[(long)s * H * FD + hd * FD + f];
  atomicAdd(&out[(long)d * FD + f], acc * (1.0f / H));
}
// l = relu((gcn+gat)/2), also emit bf16 copy for next-layer GEMMs
__global__ void combine_relu_kernel(const float* __restrict__ g1, const float* __restrict__ g2,
                                    float* __restrict__ l, __bf16* __restrict__ lb, long n) {
  long i = (long)blockIdx.x * blockDim.x + threadIdx.x;
  if (i < n) { float v = fmaxf(0.f, 0.5f * (g1[i] + g2[i])); l[i] = v; lb[i] = (__bf16)v; }
}
// A_fuse[i, 0:128]=x1 row, A_fuse[i,128:256]=x2 row (bf16)
__global__ void build_fuse_kernel(const float* __restrict__ x1, const float* __restrict__ x2,
                                  __bf16* __restrict__ out, long nF) {
  long t = (long)blockIdx.x * blockDim.x + threadIdx.x;
  if (t < nF) {
    int i = (int)(t >> 7), f = (int)(t & 127);
    out[(long)i * 256 + f]       = (__bf16)x1[t];
    out[(long)i * 256 + 128 + f] = (__bf16)x2[t];
  }
}

// ---------------------------------------------------------------------------
static void gemm(const __bf16* A, const __bf16* BT, const float* bias, float* C,
                 int M, int N, int K, hipStream_t s) {
  dim3 g(cdiv(N, 64), cdiv(M, 256));   // 4 waves/block stacked along M (64 rows each)
  wmma_gemm_nt_kernel<<<g, 128, 0, s>>>(A, BT, bias, C, M, N, K);
}

extern "C" void kernel_launch(void* const* d_in, const int* in_sizes, int n_in,
                              void* d_out, int out_size, void* d_ws, size_t ws_size,
                              hipStream_t stream) {
  // ---- inputs --------------------------------------------------------------
  const float* x_drug = (const float*)d_in[0];
  const float* x_cir  = (const float*)d_in[1];
  const int*   dei    = (const int*)d_in[2];
  const int*   cei    = (const int*)d_in[3];
  const float* ew_d   = (const float*)d_in[4];
  const float* ew_c   = (const float*)d_in[5];
  const float* Wg1_d  = (const float*)d_in[6];
  const float* bg1_d  = (const float*)d_in[7];
  const float* Wg2_d  = (const float*)d_in[8];
  const float* bg2_d  = (const float*)d_in[9];
  const float* Wa_d   = (const float*)d_in[10];
  const float* as_d   = (const float*)d_in[11];
  const float* ad_d   = (const float*)d_in[12];
  const float* We_d   = (const float*)d_in[13];
  const float* ae_d   = (const float*)d_in[14];
  const float* ba_d   = (const float*)d_in[15];
  const float* Wg1_c  = (const float*)d_in[16];
  const float* bg1_c  = (const float*)d_in[17];
  const float* Wg2_c  = (const float*)d_in[18];
  const float* bg2_c  = (const float*)d_in[19];
  const float* Wa_c   = (const float*)d_in[20];
  const float* as_c   = (const float*)d_in[21];
  const float* ad_c   = (const float*)d_in[22];
  const float* ba_c   = (const float*)d_in[23];
  const float* Kd     = (const float*)d_in[24];
  const float* bKd    = (const float*)d_in[25];
  const float* Kc     = (const float*)d_in[26];
  const float* bKc    = (const float*)d_in[27];

  const int* src_d = dei;        const int* dst_d = dei + ED;
  const int* src_c = cei;        const int* dst_c = cei + EC;

  float* out_mat  = (float*)d_out;                    // [NC, ND]
  float* drug_fea = out_mat + (size_t)NC * ND;        // [ND, OCH] (second tuple output)

  // ---- workspace carving ---------------------------------------------------
  char* ws = (char*)d_ws; size_t off = 0;
  auto alloc = [&](size_t bytes) -> void* {
    off = (off + 255) & ~(size_t)255;
    void* p = ws + off; off += bytes; return p;
  };
  float*  l1_d   = (float*) alloc(sizeof(float) * (size_t)ND * FD);
  float*  l2_d   = (float*) alloc(sizeof(float) * (size_t)ND * FD);
  float*  l1_c   = (float*) alloc(sizeof(float) * (size_t)NC * FD);
  float*  l2_c   = (float*) alloc(sizeof(float) * (size_t)NC * FD);
  float*  cirfea = (float*) alloc(sizeof(float) * (size_t)NC * OCH);
  __bf16* dfea_b = (__bf16*)alloc(sizeof(__bf16) * (size_t)ND * OCH);
  __bf16* cfea_b = (__bf16*)alloc(sizeof(__bf16) * (size_t)NC * OCH);
  __bf16* afuse  = (__bf16*)alloc(sizeof(__bf16) * (size_t)NC * 256);
  __bf16* Wg1dT  = (__bf16*)alloc(sizeof(__bf16) * FD * FD);
  __bf16* Wg2dT  = (__bf16*)alloc(sizeof(__bf16) * FD * FD);
  __bf16* WadT   = (__bf16*)alloc(sizeof(__bf16) * FD * (NH * FD));
  __bf16* Wg1cT  = (__bf16*)alloc(sizeof(__bf16) * FD * FD);
  __bf16* Wg2cT  = (__bf16*)alloc(sizeof(__bf16) * FD * FD);
  __bf16* WacT   = (__bf16*)alloc(sizeof(__bf16) * FD * FD);
  __bf16* KdB    = (__bf16*)alloc(sizeof(__bf16) * OCH * 256);
  __bf16* KcB    = (__bf16*)alloc(sizeof(__bf16) * OCH * 256);
  float*  wedot  = (float*) alloc(sizeof(float) * NH);
  float*  meanea = (float*) alloc(sizeof(float));
  // shared per-graph scratch (sized for the larger graph in each role)
  __bf16* xb     = (__bf16*)alloc(sizeof(__bf16) * (size_t)NC * FD);
  float*  hgcn   = (float*) alloc(sizeof(float) * (size_t)NC * FD);
  float*  hgat   = (float*) alloc(sizeof(float) * (size_t)ND * NH * FD); // >= NC*FD
  float*  gcn_o  = (float*) alloc(sizeof(float) * (size_t)NC * FD);
  float*  gat_o  = (float*) alloc(sizeof(float) * (size_t)NC * FD);
  float*  deg    = (float*) alloc(sizeof(float) * NC);
  float*  dinv   = (float*) alloc(sizeof(float) * NC);
  float*  asum   = (float*) alloc(sizeof(float) * ND * NH);  // >= NC*1
  float*  dsum   = (float*) alloc(sizeof(float) * ND * NH);
  float*  mbuf   = (float*) alloc(sizeof(float) * ND * NH);
  float*  ssum   = (float*) alloc(sizeof(float) * ND * NH);
  float*  alpha  = (float*) alloc(sizeof(float) * (size_t)(ED + ND) * NH); // >= EC+NC
  (void)ws_size; (void)n_in; (void)in_sizes; (void)out_size;

  const int T = 256;
  // ---- weight prep (bf16 / transposed-to-[N,K]) ----------------------------
  transpose_bf16_kernel<<<cdiv(FD*FD,T),T,0,stream>>>(Wg1_d, Wg1dT, FD, FD);
  transpose_bf16_kernel<<<cdiv(FD*FD,T),T,0,stream>>>(Wg2_d, Wg2dT, FD, FD);
  transpose_bf16_kernel<<<cdiv(FD*NH*FD,T),T,0,stream>>>(Wa_d, WadT, FD, NH*FD);
  transpose_bf16_kernel<<<cdiv(FD*FD,T),T,0,stream>>>(Wg1_c, Wg1cT, FD, FD);
  transpose_bf16_kernel<<<cdiv(FD*FD,T),T,0,stream>>>(Wg2_c, Wg2cT, FD, FD);
  transpose_bf16_kernel<<<cdiv(FD*FD,T),T,0,stream>>>(Wa_c, WacT, FD, FD);
  cvt_bf16_kernel<<<cdiv(OCH*256,T),T,0,stream>>>(Kd, KdB, OCH*256);
  cvt_bf16_kernel<<<cdiv(OCH*256,T),T,0,stream>>>(Kc, KcB, OCH*256);
  wedot_kernel<<<1,32,0,stream>>>(We_d, ae_d, wedot, NH);
  fill_kernel<<<1,32,0,stream>>>(meanea, 0.f, 1);
  mean_reduce_kernel<<<cdiv(ED,T),T,0,stream>>>(ew_d, meanea, ED, 1.0f/ED);

  // ---- one GCN+GAT layer ---------------------------------------------------
  auto layer = [&](int n, int E, const int* src, const int* dst, const float* ew,
                   const __bf16* WgT, const float* bg, const __bf16* WaT, int H,
                   const float* att_s, const float* att_d, const float* ba,
                   const float* ea, float* lout, bool first_layer) {
    // dense projections (WMMA)
    gemm(xb, WgT, nullptr, hgcn, n, FD, FD, stream);
    gemm(xb, WaT, nullptr, hgat, n, H * FD, FD, stream);
    if (first_layer) {  // degree depends only on edges: compute once per graph
      fill_kernel<<<cdiv(n,T),T,0,stream>>>(deg, 1.0f, n);
      deg_edges_kernel<<<cdiv(E,T),T,0,stream>>>(dst, ew, deg, E);
      dinv_kernel<<<cdiv(n,T),T,0,stream>>>(deg, dinv, n);
    }
    // GCN: norm scatter (norm reuses alpha scratch)
    edge_norm_kernel<<<cdiv(E,T),T,0,stream>>>(src, dst, ew, dinv, alpha, E);
    gcn_self_bias_kernel<<<cdiv((long)n*FD,T),T,0,stream>>>(hgcn, dinv, bg, gcn_o, (long)n*FD);
    gcn_scatter_kernel<<<cdiv((long)E*FD,T),T,0,stream>>>(src, dst, alpha, hgcn, gcn_o, (long)E*FD);
    // GAT: segment softmax over incoming edges (+self loops)
    att_sums_kernel<<<cdiv((long)n*H,4),128,0,stream>>>(hgat, att_s, att_d, asum, dsum, n, H);
    fill_kernel<<<cdiv((long)n*H,T),T,0,stream>>>(mbuf, -INFINITY, (long)n*H);
    fill_kernel<<<cdiv((long)n*H,T),T,0,stream>>>(ssum, 0.0f, (long)n*H);
    alpha_kernel<<<cdiv((long)(E+n)*H,T),T,0,stream>>>(src, dst, asum, dsum, ea, meanea,
                                                       wedot, alpha, mbuf, E, n, H);
    exp_kernel<<<cdiv((long)(E+n)*H,T),T,0,stream>>>(dst, mbuf, alpha, ssum, E, n, H);
    init_rows_bias_kernel<<<cdiv((long)n*FD,T),T,0,stream>>>(gat_o, ba, (long)n*FD);
    gat_scatter_kernel<<<cdiv((long)(E+n)*FD,T),T,0,stream>>>(src, dst, alpha, ssum, hgat,
                                                              gat_o, E, n, H);
    // combine, relu, and emit next-layer bf16 activations
    combine_relu_kernel<<<cdiv((long)n*FD,T),T,0,stream>>>(gcn_o, gat_o, lout, xb, (long)n*FD);
  };

  // ---- drug graph ----------------------------------------------------------
  cvt_bf16_kernel<<<cdiv((long)ND*FD,T),T,0,stream>>>(x_drug, xb, (long)ND*FD);
  layer(ND, ED, src_d, dst_d, ew_d, Wg1dT, bg1_d, WadT, NH, as_d, ad_d, ba_d, ew_d, l1_d, true);
  layer(ND, ED, src_d, dst_d, ew_d, Wg2dT, bg2_d, WadT, NH, as_d, ad_d, ba_d, ew_d, l2_d, false);

  // ---- circRNA graph -------------------------------------------------------
  cvt_bf16_kernel<<<cdiv((long)NC*FD,T),T,0,stream>>>(x_cir, xb, (long)NC*FD);
  layer(NC, EC, src_c, dst_c, ew_c, Wg1cT, bg1_c, WacT, 1, as_c, ad_c, ba_c, nullptr, l1_c, true);
  layer(NC, EC, src_c, dst_c, ew_c, Wg2cT, bg2_c, WacT, 1, as_c, ad_c, ba_c, nullptr, l2_c, false);

  // ---- CNN fuse (GEMM over concat features) --------------------------------
  build_fuse_kernel<<<cdiv((long)ND*FD,T),T,0,stream>>>(l1_d, l2_d, afuse, (long)ND*FD);
  gemm(afuse, KdB, bKd, drug_fea, ND, OCH, 256, stream);        // writes d_out tail directly
  build_fuse_kernel<<<cdiv((long)NC*FD,T),T,0,stream>>>(l1_c, l2_c, afuse, (long)NC*FD);
  gemm(afuse, KcB, bKc, cirfea, NC, OCH, 256, stream);

  // ---- final big GEMM: cir_fea @ drug_fea^T --------------------------------
  cvt_bf16_kernel<<<cdiv((long)ND*OCH,T),T,0,stream>>>(drug_fea, dfea_b, (long)ND*OCH);
  cvt_bf16_kernel<<<cdiv((long)NC*OCH,T),T,0,stream>>>(cirfea,  cfea_b, (long)NC*OCH);
  gemm(cfea_b, dfea_b, nullptr, out_mat, NC, ND, OCH, stream);  // [16000 x 8000], store-BW bound
}
